// MixtralSparseMoeBlock_11948599018371
// MI455X (gfx1250) — matrix operand order repaired and checked by reading, not compiled, not run on hardware
//
#include <hip/hip_runtime.h>

// ---------------------------------------------------------------------------
// MI455X (gfx1250) fused Mixtral MoE block.
//   B=4 S=4096 H=1024 I=3584 E=8 K=2  -> T=16384 tokens.
// Compute-bound on bf16 WMMA (~7.2e11 useful FLOPs, top-2 only); ~0.3 GB min
// traffic with bf16 mirrors. Pipeline:
//   0) fp32 -> bf16 mirrors; w1/w3 stored transposed [e][i][h], w2 transposed
//      [e][h][i] so every expert-kernel LDS tile is contiguous in global
//   1) router (wave/token): softmax top-2 renorm + per-expert token lists
//   2) fused expert FFN: all LDS staging via GLOBAL_LOAD_ASYNC_TO_LDS_B128;
//      split s_wait_asynccnt so stage-B operand DMA overlaps stage-A WMMAs;
//      bf16 WMMA stage-A (x@w1, x@w3), SwiGLU + routing weight folded,
//      stage-B (hidden@w2) register-accumulated over all of I, fp32 atomic
//      scatter-add.
// ---------------------------------------------------------------------------

#if __has_builtin(__builtin_amdgcn_global_load_async_to_lds_b128)
#define HAVE_ASYNC_B128 1
#else
#define HAVE_ASYNC_B128 0
#endif
#if __has_builtin(__builtin_amdgcn_s_wait_asynccnt)
#define HAVE_WAIT_ASYNC 1
#else
#define HAVE_WAIT_ASYNC 0
#endif
#define HAVE_ASYNC (HAVE_ASYNC_B128 && HAVE_WAIT_ASYNC)

typedef __bf16 v16bf __attribute__((ext_vector_type(16)));
typedef float  v8f   __attribute__((ext_vector_type(8)));
typedef int    v4i   __attribute__((ext_vector_type(4)));

constexpr int T = 16384;         // tokens
constexpr int H = 1024;
constexpr int I = 3584;
constexpr int E = 8;

constexpr int TM = 32;           // token tile per block
constexpr int CI = 32;           // I-chunk (== WMMA K for bf16)
constexpr int NC = I / CI;       // 112 chunks

// LDS element strides (padded; strides stay multiples of 16B for async b128)
constexpr int XK = H + 8;        // 1032 bf16 -> 2064 B rows
constexpr int WK = H + 8;        // 1032 bf16 -> 2064 B rows
constexpr int HS = 33;           // f32 hidden scratch row stride
constexpr int HK = 40;           // bf16 hidden / w2 k-stride -> 80 B rows

// LDS byte layout
constexpr int OFF_TOK  = 0;                        // 32 int
constexpr int OFF_WROW = OFF_TOK  + TM * 4;        // 32 f32
constexpr int OFF_XA   = OFF_WROW + TM * 4 + 128;  // TM x XK bf16 (16B aligned)
constexpr int OFF_WB1  = OFF_XA   + TM * XK * 2;   // CI x WK bf16 ([n][k])
constexpr int OFF_WB3  = OFF_WB1  + CI * WK * 2;
constexpr int OFF_HS1  = OFF_WB3  + CI * WK * 2;   // TM x HS f32
constexpr int OFF_HS3  = OFF_HS1  + TM * HS * 4;
constexpr int OFF_HT   = OFF_HS3  + TM * HS * 4;   // TM x HK bf16 ([m][k])
constexpr int OFF_W2B  = OFF_HT   + TM * HK * 2 + 48; // H x HK bf16 ([n][k]), 16B aligned
constexpr int SMEM_BYTES = OFF_W2B + H * HK * 2;   // ~284.7 KB (< 320 KB WGP LDS)

// workspace byte layout (assumes ws_size >= ~210 MB)
constexpr size_t WS_COUNTS = 0;                               // E int
constexpr size_t WS_LIST   = 256;                             // E*T int
constexpr size_t WS_WFULL  = WS_LIST + (size_t)E * T * 4;     // T*E f32
constexpr size_t WS_XBF    = 2u << 20;                        // T*H bf16
constexpr size_t WS_W1T    = WS_XBF + (size_t)T * H * 2;      // E*I*H bf16 [e][i][h]
constexpr size_t WS_W3T    = WS_W1T + (size_t)E * H * I * 2;  // E*I*H bf16 [e][i][h]
constexpr size_t WS_W2T    = WS_W3T + (size_t)E * H * I * 2;  // E*H*I bf16 [e][h][i]

// fp32 -> bf16, round to nearest even
__device__ inline unsigned short f2bf(float f) {
  unsigned int u = __float_as_uint(f);
  unsigned int r = u + 0x7FFFu + ((u >> 16) & 1u);
  return (unsigned short)(r >> 16);
}

// 16-byte global -> LDS copy (async on CDNA5; ASYNCcnt-tracked)
__device__ inline void copy16_g2l(const void* g, void* l) {
#if HAVE_ASYNC
  __builtin_amdgcn_global_load_async_to_lds_b128(
      (__attribute__((address_space(1))) v4i*)g,
      (__attribute__((address_space(3))) v4i*)l, 0, 0);
#else
  *(uint4*)l = *(const uint4*)g;
#endif
}

// A-operand fragment, 16x32 bf16 (ISA 7.12.2): LDS layout [m][k].
__device__ inline v16bf load_a_frag(const unsigned short* base, int rowStride) {
  int lane = threadIdx.x & 31;
  const unsigned short* rp = base + (lane & 15) * rowStride + (lane >> 4) * 16;
  union { v16bf v; unsigned int u[8]; } f;
#pragma unroll
  for (int p = 0; p < 8; ++p) {
    int k = 2 * p;
    f.u[p] = *(const unsigned int*)(rp + k);
  }
  return f.v;
}

// B-operand fragment, 32x16 bf16: LDS layout transposed [n][k].
__device__ inline v16bf load_b_frag(const unsigned short* base, int rowStride) {
  int lane = threadIdx.x & 31;
  const unsigned short* rp = base + (lane & 15) * rowStride + (lane >> 4) * 16;
  union { v16bf v; unsigned int u[8]; } f;
#pragma unroll
  for (int p = 0; p < 8; ++p)
    f.u[p] = *(const unsigned int*)(rp + 2 * p);
  return f.v;
}

// ---------------------------------------------------------------------------
// fp32 -> bf16 bulk convert (no transpose): 8 elements/thread.
// ---------------------------------------------------------------------------
__global__ __launch_bounds__(256) void cvt_bf16_kernel(
    const float* __restrict__ src, unsigned short* __restrict__ dst, int n8) {
  int i = blockIdx.x * 256 + threadIdx.x;
  if (i >= n8) return;
  const float4* s = (const float4*)src + (size_t)i * 2;
  float4 a = s[0], b = s[1];
  uint4 o;
  o.x = (unsigned)f2bf(a.x) | ((unsigned)f2bf(a.y) << 16);
  o.y = (unsigned)f2bf(a.z) | ((unsigned)f2bf(a.w) << 16);
  o.z = (unsigned)f2bf(b.x) | ((unsigned)f2bf(b.y) << 16);
  o.w = (unsigned)f2bf(b.z) | ((unsigned)f2bf(b.w) << 16);
  ((uint4*)dst)[i] = o;
}

// ---------------------------------------------------------------------------
// fp32 [e][R][C] -> bf16 [e][C][R] tiled transpose+convert (32x32 LDS tiles).
// ---------------------------------------------------------------------------
__global__ __launch_bounds__(256) void cvt_transpose_kernel(
    const float* __restrict__ src, unsigned short* __restrict__ dst,
    int R, int C) {
  __shared__ float tile[32][33];
  const int e  = blockIdx.z;
  const int c0 = blockIdx.x * 32, r0 = blockIdx.y * 32;
  const float*          se = src + (size_t)e * R * C;
  unsigned short*       de = dst + (size_t)e * R * C;
  const int tx = threadIdx.x, ty = threadIdx.y;
#pragma unroll
  for (int i = ty; i < 32; i += 8)
    tile[i][tx] = se[(size_t)(r0 + i) * C + c0 + tx];
  __syncthreads();
#pragma unroll
  for (int i = ty; i < 32; i += 8)
    de[(size_t)(c0 + i) * R + r0 + tx] = f2bf(tile[tx][i]);
}

// ---------------------------------------------------------------------------
// Router: one wave per token; softmax/top-2/renorm; builds expert lists.
// ---------------------------------------------------------------------------
__global__ __launch_bounds__(256) void router_kernel(
    const float* __restrict__ x, const float* __restrict__ gate,
    float* __restrict__ wfull, int* __restrict__ counts, int* __restrict__ list) {
  const int lane = threadIdx.x & 31;
  const int wv   = threadIdx.x >> 5;
  const int t    = blockIdx.x * 8 + wv;

  float acc[E];
#pragma unroll
  for (int e = 0; e < E; ++e) acc[e] = 0.0f;

  const float* xr = x + (size_t)t * H;
  for (int h = lane; h < H; h += 32) {
    float xv = xr[h];
#pragma unroll
    for (int e = 0; e < E; ++e) acc[e] += xv * gate[e * H + h];
  }
#pragma unroll
  for (int e = 0; e < E; ++e)
    for (int off = 16; off > 0; off >>= 1)
      acc[e] += __shfl_xor(acc[e], off, 32);

  if (lane == 0) {
    float m = acc[0];
#pragma unroll
    for (int e = 1; e < E; ++e) m = fmaxf(m, acc[e]);
    float p[E];
#pragma unroll
    for (int e = 0; e < E; ++e) p[e] = __expf(acc[e] - m);
    int i1 = 0;
#pragma unroll
    for (int e = 1; e < E; ++e) if (p[e] > p[i1]) i1 = e;
    int i2 = (i1 == 0) ? 1 : 0;
#pragma unroll
    for (int e = 0; e < E; ++e) if (e != i1 && p[e] > p[i2]) i2 = e;
    float denom = p[i1] + p[i2];
#pragma unroll
    for (int e = 0; e < E; ++e) wfull[(size_t)t * E + e] = 0.0f;
    wfull[(size_t)t * E + i1] = p[i1] / denom;
    wfull[(size_t)t * E + i2] = p[i2] / denom;
    int p1 = atomicAdd(&counts[i1], 1);
    list[i1 * T + p1] = t;
    int p2 = atomicAdd(&counts[i2], 1);
    list[i2 * T + p2] = t;
  }
}

// ---------------------------------------------------------------------------
// Fused expert FFN (bf16 operands, pre-transposed weights).
// grid = (E, T/TM); 256 threads = 8 waves.
// Per chunk, each wave issues (in order) 32 WB async copies then 16 W2B async
// copies; s_wait_asynccnt 16 releases stage A while W2B still streams, and
// asynccnt 0 is only drained right before stage B consumes W2B.
// ---------------------------------------------------------------------------
__global__ __launch_bounds__(256) void moe_expert_kernel(
    const unsigned short* __restrict__ xbf,  const unsigned short* __restrict__ w1t,
    const unsigned short* __restrict__ w3t,  const unsigned short* __restrict__ w2t,
    const float* __restrict__ wfull, const int* __restrict__ counts,
    const int* __restrict__ list, float* __restrict__ out) {
  extern __shared__ char smem[];
  int*            tokS  = (int*)(smem + OFF_TOK);
  float*          wrowS = (float*)(smem + OFF_WROW);
  unsigned short* XA    = (unsigned short*)(smem + OFF_XA);
  unsigned short* WB1   = (unsigned short*)(smem + OFF_WB1);
  unsigned short* WB3   = (unsigned short*)(smem + OFF_WB3);
  float*          Hs1   = (float*)(smem + OFF_HS1);
  float*          Hs3   = (float*)(smem + OFF_HS3);
  unsigned short* HT    = (unsigned short*)(smem + OFF_HT);
  unsigned short* W2B   = (unsigned short*)(smem + OFF_W2B);

  const int e   = blockIdx.x;
  const int m0  = blockIdx.y * TM;
  const int cnt = counts[e];
  if (m0 >= cnt) return;          // uniform early-exit

  const int tid  = threadIdx.x;
  const int wv   = tid >> 5;
  const int lane = tid & 31;

  if (tid < TM) {                  // gather token ids + folded routing weights
    int row = m0 + tid;
    int cl  = (row < cnt) ? row : (cnt - 1);
    int tok = list[e * T + cl];
    tokS[tid]  = tok;
    wrowS[tid] = (row < cnt) ? wfull[(size_t)tok * E + e] : 0.0f;
  }
  __syncthreads();

  // ---- stage gathered X tile (async, 128 x 16B units per row) ------------
  for (int u = tid; u < TM * (H / 8); u += 256) {
    int r = u >> 7, o = (u & 127) << 4;
    copy16_g2l((const char*)(xbf + (size_t)tokS[r] * H) + o,
               (char*)XA + r * (XK * 2) + o);
  }
#if HAVE_ASYNC
  __builtin_amdgcn_s_wait_asynccnt(0);
#endif

  const v8f vzero = {0.f, 0.f, 0.f, 0.f, 0.f, 0.f, 0.f, 0.f};
  v8f acc[2][8];
#pragma unroll
  for (int a = 0; a < 2; ++a)
#pragma unroll
    for (int b = 0; b < 8; ++b) acc[a][b] = vzero;

  const int mat = wv & 1, mt = (wv >> 1) & 1, nt = (wv >> 2) & 1;

  const unsigned short* w1e = w1t + (size_t)e * H * I;   // [i][h]
  const unsigned short* w3e = w3t + (size_t)e * H * I;   // [i][h]
  const unsigned short* w2e = w2t + (size_t)e * I * H;   // [h][i]

  __syncthreads();                 // XA ready

  for (int c = 0; c < NC; ++c) {
    const int c0 = c * CI;
    if (c + 1 < NC) {              // hint next weight chunk toward L2
      __builtin_prefetch(w1e + (size_t)(c0 + CI) * H, 0, 1);
      __builtin_prefetch(w3e + (size_t)(c0 + CI) * H, 0, 1);
      __builtin_prefetch(w2e + (size_t)(c0 + CI), 0, 1);
    }

    // (1) w1t/w3t rows i=c0..c0+31 (2 KB contiguous each) -> WB1/WB3 [n][k]
    //     32 async instructions per wave, issued first
    for (int u = tid; u < 2 * CI * (H / 8); u += 256) {
      int m_ = u >> 12;                       // 0: w1, 1: w3
      int v_ = u & 4095;
      int n = v_ >> 7, o = (v_ & 127) << 4;
      const unsigned short* src = (m_ ? w3e : w1e) + (size_t)(c0 + n) * H;
      unsigned short* dstl = (m_ ? WB3 : WB1) + n * WK;
      copy16_g2l((const char*)src + o, (char*)dstl + o);
    }
    // (2) w2t 64B row segments [h][c0..c0+31] -> W2B [n=h][k]
    //     16 async instructions per wave, issued second
    for (int u = tid; u < H * 4; u += 256) {
      int h = u >> 2, o = (u & 3) << 4;
      copy16_g2l((const char*)(w2e + (size_t)h * I + c0) + o,
                 (char*)(W2B + h * HK) + o);
    }
#if HAVE_ASYNC
    // async loads complete in order: cnt<=16 means all 32 WB copies landed;
    // the 16 W2B copies keep streaming underneath stage A + SwiGLU.
    __builtin_amdgcn_s_wait_asynccnt(16);
#endif
    __syncthreads();

    // ---- stage A: 32x32 hidden-chunk GEMM (16x16 per wave), K = H = 1024 -------------------
    {
      const unsigned short* wb = mat ? WB3 : WB1;
      v8f hacc = vzero;
#pragma unroll 8
      for (int kk = 0; kk < H / 32; ++kk) {
        v16bf af = load_a_frag(XA + mt * 16 * XK + kk * 32, XK);
        v16bf bf = load_b_frag(wb + nt * 16 * WK + kk * 32, WK);
        hacc = __builtin_amdgcn_wmma_f32_16x16x32_bf16(
            false, af, false, bf, (short)0, hacc, false, false);
      }
      float* hs = mat ? Hs3 : Hs1;
      int col = lane & 15, hb = lane >> 4;
#pragma unroll
      for (int r = 0; r < 8; ++r)
        hs[(mt * 16 + hb * 8 + r) * HS + nt * 16 + col] = hacc[r];
    }
    __syncthreads();

    // ---- SwiGLU + routing weight, pack bf16 A-operand ---------------------
    for (int j = tid; j < TM * CI; j += 256) {
      int m = j >> 5, nn = j & (CI - 1);
      float h1v = Hs1[m * HS + nn];
      float h3v = Hs3[m * HS + nn];
      float sv  = h1v / (1.0f + __expf(-h1v));
      HT[m * HK + nn] = f2bf(sv * h3v * wrowS[m]);
    }
#if HAVE_ASYNC
    __builtin_amdgcn_s_wait_asynccnt(0);   // W2B fully landed
#endif
    __syncthreads();

    // ---- stage B: acc += hidden_chunk @ w2_chunk; wave owns 128 H-cols ----
#pragma unroll
    for (int nt2 = 0; nt2 < 8; ++nt2) {
      v16bf bf = load_b_frag(W2B + (wv * 128 + nt2 * 16) * HK, HK);
#pragma unroll
      for (int mt2 = 0; mt2 < 2; ++mt2) {
        v16bf af = load_a_frag(HT + mt2 * 16 * HK, HK);
        acc[mt2][nt2] = __builtin_amdgcn_wmma_f32_16x16x32_bf16(
            false, af, false, bf, (short)0, acc[mt2][nt2], false, false);
      }
    }
    __syncthreads();
  }

  // scatter-add (routing weights already folded; padded rows carry zero)
  const int col = lane & 15, hb = lane >> 4;
#pragma unroll
  for (int mt2 = 0; mt2 < 2; ++mt2)
#pragma unroll
    for (int nt2 = 0; nt2 < 8; ++nt2)
#pragma unroll
      for (int r = 0; r < 8; ++r) {
        int m = mt2 * 16 + hb * 8 + r;
        atomicAdd(out + (size_t)tokS[m] * H + wv * 128 + nt2 * 16 + col,
                  acc[mt2][nt2][r]);
      }
}

// ---------------------------------------------------------------------------
extern "C" void kernel_launch(void* const* d_in, const int* in_sizes, int n_in,
                              void* d_out, int out_size, void* d_ws, size_t ws_size,
                              hipStream_t stream) {
  const float* x    = (const float*)d_in[0];
  const float* gate = (const float*)d_in[1];
  const float* w1   = (const float*)d_in[2];
  const float* w3   = (const float*)d_in[3];
  const float* w2   = (const float*)d_in[4];
  float* out = (float*)d_out;

  char* ws = (char*)d_ws;
  int*   counts = (int*)(ws + WS_COUNTS);
  int*   list   = (int*)(ws + WS_LIST);
  float* wfull  = (float*)(ws + WS_WFULL);
  unsigned short* xbf = (unsigned short*)(ws + WS_XBF);
  unsigned short* w1t = (unsigned short*)(ws + WS_W1T);
  unsigned short* w3t = (unsigned short*)(ws + WS_W3T);
  unsigned short* w2t = (unsigned short*)(ws + WS_W2T);

  (void)hipMemsetAsync(counts, 0, E * sizeof(int), stream);
  (void)hipMemsetAsync(out, 0, (size_t)out_size * sizeof(float), stream);

  // bf16 mirrors (recomputed every call: deterministic, no cross-call state)
  {
    int nx = (T * H) / 8;
    cvt_bf16_kernel<<<(nx + 255) / 256, 256, 0, stream>>>(x, xbf, nx);
    dim3 blk(32, 8);
    cvt_transpose_kernel<<<dim3(I / 32, H / 32, E), blk, 0, stream>>>(w1, w1t, H, I);
    cvt_transpose_kernel<<<dim3(I / 32, H / 32, E), blk, 0, stream>>>(w3, w3t, H, I);
    cvt_transpose_kernel<<<dim3(H / 32, I / 32, E), blk, 0, stream>>>(w2, w2t, I, H);
  }

  router_kernel<<<T / 8, 256, 0, stream>>>(x, gate, wfull, counts, list);

  dim3 grid(E, T / TM);
  moe_expert_kernel<<<grid, 256, SMEM_BYTES, stream>>>(
      xbf, w1t, w3t, w2t, wfull, counts, list, out);
}
